// DotProductAttention_83159156785607
// MI455X (gfx1250) — matrix operand order repaired and checked by reading, not compiled, not run on hardware
//
#include <hip/hip_runtime.h>

// ---------------------------------------------------------------------------
// Fused batched attention for gfx1250 (MI455X), wave32 + v_wmma_f32_16x16x32_f16
//   score = (Q.K^T + (mask-1)*1e4) / sqrt(512);  attn = softmax(score);  ctx = attn.V
// Outputs (concatenated in d_out): context [8,2048,512] fp32, attn [8,2048,2048] fp32
// 32 q-rows per workgroup: each converted K/V fragment feeds 2 WMMAs, halving
// the v_cvt_pk_f16_f32 : v_wmma ratio that bounds the inner loops.
// ---------------------------------------------------------------------------

typedef __attribute__((ext_vector_type(16))) _Float16 v16h;
typedef __attribute__((ext_vector_type(8)))  _Float16 v8h;
typedef __attribute__((ext_vector_type(4)))  _Float16 v4h;
typedef __attribute__((ext_vector_type(8)))  float    v8f;
typedef __attribute__((ext_vector_type(4)))  float    v4f;

#define B_   8
#define LQ_  2048
#define LK_  2048
#define D_   512
#define QT   32     // q rows per workgroup
#define NW   8      // waves per workgroup
#define NT   256    // threads per workgroup

static __device__ __forceinline__ v16h cat16(v8h a, v8h b) {
  return __builtin_shufflevector(a, b, 0,1,2,3,4,5,6,7,8,9,10,11,12,13,14,15);
}

static __device__ __forceinline__ v8h cvt8(v4f a, v4f b) {
  v8h h;
  h[0]=(_Float16)a[0]; h[1]=(_Float16)a[1]; h[2]=(_Float16)a[2]; h[3]=(_Float16)a[3];
  h[4]=(_Float16)b[0]; h[5]=(_Float16)b[1]; h[6]=(_Float16)b[2]; h[7]=(_Float16)b[3];
  return h;
}

__global__ __launch_bounds__(NT)
void attn_fused_f16wmma(const float* __restrict__ Q,
                        const float* __restrict__ K,
                        const float* __restrict__ V,
                        const float* __restrict__ M,
                        float* __restrict__ ctx,
                        float* __restrict__ attn)
{
  extern __shared__ char smem[];
  _Float16* sc = (_Float16*)smem;             // [32][2048] f16 scores -> probs (128 KB)
  _Float16* qs = (_Float16*)(smem + 131072);  // [32][512]  f16 Q tile (32 KB), phase 1 only
  _Float16* vt = (_Float16*)(smem + 131072);  // [512][32]  f16 V^T chunk (32 KB), phase 3

  const int tid  = threadIdx.x;
  const int w    = tid >> 5;
  const int lane = tid & 31;
  const int l16  = lane & 15;
  const int hi   = lane >> 4;              // 0 or 1 (half-wave)
  const int b    = blockIdx.x >> 6;        // batch (64 WGs per batch)
  const int q0   = (blockIdx.x & 63) << 5; // 32-row q stripe

  // ---------------- phase 0: stage Q tile (fp32 -> f16 in LDS) ----------------
  {
    const int r  = tid >> 3;               // 0..31
    const int c0 = (tid & 7) << 6;         // 0..448, 64 floats each
    const float* src = Q + ((size_t)(b * LQ_ + q0 + r) * D_ + c0);
    _Float16* dst = qs + r * D_ + c0;
    #pragma unroll
    for (int i = 0; i < 8; ++i) {
      v4f f0 = *(const v4f*)(src + i * 8);
      v4f f1 = *(const v4f*)(src + i * 8 + 4);
      *(v8h*)(dst + i * 8) = cvt8(f0, f1);
    }
  }
  __syncthreads();

  // ---------------- phase 1: S = Q.K^T + bias, scaled, into LDS (f16) ---------
  const float inv_sqrt_d = 0.04419417382415922f;   // 1/sqrt(512)
  for (int t = w; t < LK_ / 16; t += NW) {
    const int k0 = t << 4;
    v8f acc0 = {0.f,0.f,0.f,0.f,0.f,0.f,0.f,0.f};  // q rows 0..15
    v8f acc1 = {0.f,0.f,0.f,0.f,0.f,0.f,0.f,0.f};  // q rows 16..31
    // B fragment: lane holds key column k0+l16, K-dims {hi*8..hi*8+7, hi*8+16..hi*8+23} per chunk
    const float* kb = K + ((size_t)(b * LK_ + k0 + l16) * D_) + (hi << 3);
    // A fragments: lane holds Q row l16 (and l16+16), dims {off..off+7, off+16..off+23}
    const _Float16* qa = qs + l16 * D_ + (hi << 3);
    const _Float16* qc = qs + (l16 + 16) * D_ + (hi << 3);
    #pragma unroll 4
    for (int c = 0; c < D_; c += 32) {
      v4f f0 = *(const v4f*)(kb + c);
      v4f f1 = *(const v4f*)(kb + c + 4);
      v4f f2 = *(const v4f*)(kb + c + 16);
      v4f f3 = *(const v4f*)(kb + c + 20);
      const v16h bf = cat16(cvt8(f0, f1), cvt8(f2, f3));
      v8h a0 = *(const v8h*)(qa + c);
      v8h a1 = *(const v8h*)(qa + c + 16);
      acc0 = __builtin_amdgcn_wmma_f32_16x16x32_f16(
               false, cat16(a0, a1), false, bf, (short)0, acc0, false, false);
      v8h a2 = *(const v8h*)(qc + c);
      v8h a3 = *(const v8h*)(qc + c + 16);
      acc1 = __builtin_amdgcn_wmma_f32_16x16x32_f16(
               false, cat16(a2, a3), false, bf, (short)0, acc1, false, false);
    }
    // mask bias applied BEFORE 1/sqrt(d), matching the reference
    const float bias = (M[(size_t)b * LK_ + k0 + l16] - 1.f) * 10000.f;
    #pragma unroll
    for (int v = 0; v < 8; ++v) {
      sc[((hi << 3) + v)      * LK_ + k0 + l16] = (_Float16)((acc0[v] + bias) * inv_sqrt_d);
      sc[((hi << 3) + v + 16) * LK_ + k0 + l16] = (_Float16)((acc1[v] + bias) * inv_sqrt_d);
    }
  }
  __syncthreads();

  // ---------------- phase 2: row softmax; write attn (fp32) + probs (f16 LDS) --
  float* attnb = attn + ((size_t)b * LQ_ + q0) * (size_t)LK_;
  #pragma unroll
  for (int rr = 0; rr < 4; ++rr) {
    const int r = w * 4 + rr;
    _Float16* row = sc + (size_t)r * LK_;
    float e[64];
    float mx = -3.0e38f;
    #pragma unroll
    for (int i = 0; i < 64; ++i) { e[i] = (float)row[lane + 32 * i]; mx = fmaxf(mx, e[i]); }
    #pragma unroll
    for (int s = 16; s > 0; s >>= 1) mx = fmaxf(mx, __shfl_xor(mx, s, 32));
    float sum = 0.f;
    #pragma unroll
    for (int i = 0; i < 64; ++i) { e[i] = __expf(e[i] - mx); sum += e[i]; }
    #pragma unroll
    for (int s = 16; s > 0; s >>= 1) sum += __shfl_xor(sum, s, 32);
    const float isum = 1.f / sum;
    float* arow = attnb + (size_t)r * LK_;
    #pragma unroll
    for (int i = 0; i < 64; ++i) {
      const float p = e[i] * isum;
      arow[lane + 32 * i] = p;               // coalesced fp32 attn output
      row[lane + 32 * i] = (_Float16)p;      // f16 probs for GEMM2
    }
  }
  __syncthreads();

  // ---------------- phase 3: ctx = P.V  (stage V^T chunks in LDS) -------------
  v8f o0[4], o1[4];
  #pragma unroll
  for (int t = 0; t < 4; ++t)
    #pragma unroll
    for (int v = 0; v < 8; ++v) { o0[t][v] = 0.f; o1[t][v] = 0.f; }

  for (int kc = 0; kc < LK_; kc += 32) {
    // cooperatively stage vt[n][k] = (f16)V[b][kc+k][n], n=0..511, k=0..31
    // 4 keys packed per ds_store_b64; global reads row-coalesced across the WG
    #pragma unroll
    for (int it = 0; it < 4; ++it) {
      const int linear = it * NT + tid;      // 0..1023
      const int ng = linear & 127;           // dim group of 4
      const int kq = linear >> 7;            // key quad 0..7
      const float* vp = V + ((size_t)(b * LK_ + kc + kq * 4) * D_) + ng * 4;
      v4f r0 = *(const v4f*)vp;
      v4f r1 = *(const v4f*)(vp + D_);
      v4f r2 = *(const v4f*)(vp + 2 * D_);
      v4f r3 = *(const v4f*)(vp + 3 * D_);
      #pragma unroll
      for (int j = 0; j < 4; ++j) {
        v4h p;
        p[0] = (_Float16)r0[j]; p[1] = (_Float16)r1[j];
        p[2] = (_Float16)r2[j]; p[3] = (_Float16)r3[j];
        *(v4h*)(vt + (ng * 4 + j) * 32 + kq * 4) = p;
      }
    }
    __syncthreads();
    // A fragments: probs rows l16 and l16+16, keys {off..off+7, off+16..off+23}
    const int off = kc + (hi << 3);
    v8h a0 = *(const v8h*)(sc + l16 * LK_ + off);
    v8h a1 = *(const v8h*)(sc + l16 * LK_ + off + 16);
    v8h a2 = *(const v8h*)(sc + (l16 + 16) * LK_ + off);
    v8h a3 = *(const v8h*)(sc + (l16 + 16) * LK_ + off + 16);
    const v16h af0 = cat16(a0, a1);
    const v16h af1 = cat16(a2, a3);
    #pragma unroll
    for (int t = 0; t < 4; ++t) {
      const int n0 = w * 64 + t * 16;
      // B fragment: lane holds output-dim column n0+l16, key rows {hi*8..hi*8+7, hi*8+16..hi*8+23}
      const _Float16* vb = vt + (n0 + l16) * 32 + (hi << 3);
      v8h b0 = *(const v8h*)vb;
      v8h b1 = *(const v8h*)(vb + 16);
      const v16h bf = cat16(b0, b1);
      o0[t] = __builtin_amdgcn_wmma_f32_16x16x32_f16(
                false, af0, false, bf, (short)0, o0[t], false, false);
      o1[t] = __builtin_amdgcn_wmma_f32_16x16x32_f16(
                false, af1, false, bf, (short)0, o1[t], false, false);
    }
    __syncthreads();
  }

  // ---------------- epilogue: write context (fp32, coalesced per 16 lanes) ----
  #pragma unroll
  for (int t = 0; t < 4; ++t) {
    const int n = w * 64 + t * 16 + l16;
    #pragma unroll
    for (int v = 0; v < 8; ++v) {
      ctx[((size_t)(b * LQ_ + q0 + (hi << 3) + v))      * D_ + n] = o0[t][v];
      ctx[((size_t)(b * LQ_ + q0 + (hi << 3) + v + 16)) * D_ + n] = o1[t][v];
    }
  }
}

extern "C" void kernel_launch(void* const* d_in, const int* in_sizes, int n_in,
                              void* d_out, int out_size, void* d_ws, size_t ws_size,
                              hipStream_t stream) {
  const float* Q  = (const float*)d_in[0];
  const float* K  = (const float*)d_in[1];
  const float* V  = (const float*)d_in[2];
  const float* M  = (const float*)d_in[3];
  float* ctx  = (float*)d_out;
  float* attn = (float*)d_out + (size_t)B_ * LQ_ * D_;

  const int smem_bytes = 131072 + 32768;  // scores(128K) + max(Qtile 32K, V^T 32K)
  (void)hipFuncSetAttribute(reinterpret_cast<const void*>(attn_fused_f16wmma),
                            hipFuncAttributeMaxDynamicSharedMemorySize, smem_bytes);

  dim3 grid(B_ * (LQ_ / QT));             // 512 workgroups; same-batch blocks adjacent for L2 reuse
  attn_fused_f16wmma<<<grid, NT, smem_bytes, stream>>>(Q, K, V, M, ctx, attn);
}
